// CTRAttention_91053306675422
// MI455X (gfx1250) — matrix operand order repaired and checked by reading, not compiled, not run on hardware
//
#include <hip/hip_runtime.h>

// ---------- types ----------
typedef __bf16 bf16_t;
typedef bf16_t bf16x16 __attribute__((ext_vector_type(16)));
typedef float  v8f     __attribute__((ext_vector_type(8)));
typedef unsigned u32x4 __attribute__((ext_vector_type(4)));
typedef float  f32x4   __attribute__((ext_vector_type(4)));

// ---------- problem constants ----------
#define BATCH      16
#define TFRAMES    243
#define JJ         17
#define DIMC       512
#define INTERC     256
#define QKVC       768
#define NHEADS     8
#define SCALE_F    0.17677669529663687f   // 32^-0.5
#define NROWS      (BATCH * TFRAMES * JJ) // 66096
#define MTILES     (NROWS / 16)           // 4131

// ---------- bf16 split helpers ----------
__device__ __forceinline__ bf16_t f2bf_rne(float x) {
  unsigned u = __builtin_bit_cast(unsigned, x);
  unsigned r = u + 0x7FFFu + ((u >> 16) & 1u);
  unsigned short hs = (unsigned short)(r >> 16);
  return __builtin_bit_cast(bf16_t, hs);
}
__device__ __forceinline__ float bf2f(bf16_t b) {
  unsigned short hs = __builtin_bit_cast(unsigned short, b);
  unsigned u = ((unsigned)hs) << 16;
  return __builtin_bit_cast(float, u);
}

// ---------- fragment loads ----------
union FragU { u32x4 q[2]; bf16x16 v; };

// B fragment (pre-converted bf16, row-major [Ncols, K]): 16 contiguous K values
__device__ __forceinline__ bf16x16 load_b_frag(const bf16_t* p) {
  FragU u;
  u.q[0] = *reinterpret_cast<const u32x4*>(p);
  u.q[1] = *reinterpret_cast<const u32x4*>(p + 8);
  return u.v;
}

// A fragment from fp32, split hi/lo on the fly.
// elems 0..15 at p[0..15]
__device__ __forceinline__ void cvt_a_frag(const float* p, bf16x16& hi, bf16x16& lo) {
  f32x4 a0 = *reinterpret_cast<const f32x4*>(p);
  f32x4 a1 = *reinterpret_cast<const f32x4*>(p + 4);
  f32x4 a2 = *reinterpret_cast<const f32x4*>(p + 8);
  f32x4 a3 = *reinterpret_cast<const f32x4*>(p + 12);
  float f[16];
#pragma unroll
  for (int i = 0; i < 4; ++i) { f[i] = a0[i]; f[4+i] = a1[i]; f[8+i] = a2[i]; f[12+i] = a3[i]; }
#pragma unroll
  for (int i = 0; i < 16; ++i) {
    bf16_t h = f2bf_rne(f[i]);
    hi[i] = h;
    lo[i] = f2bf_rne(f[i] - bf2f(h));
  }
}

// ---------- weight split kernel (tiny) ----------
__global__ void convert_hilo(const float* __restrict__ src, bf16_t* __restrict__ hi,
                             bf16_t* __restrict__ lo, int n) {
  int i = blockIdx.x * blockDim.x + threadIdx.x;
  int stride = gridDim.x * blockDim.x;
  for (; i < n; i += stride) {
    float v = src[i];
    bf16_t h = f2bf_rne(v);
    hi[i] = h;
    lo[i] = f2bf_rne(v - bf2f(h));
  }
}

// ---------- GEMM: C[M,Ncols] = A[M,K](fp32, split on the fly) * B^T (B[Ncols,K] bf16 hi/lo) + bias
// One wave -> 16 rows x 128 cols (8 n-tiles). Block = 8 waves (8 m-tiles). grid = (Ncols/128, ceil(M/128))
template<int K>
__launch_bounds__(256)
__global__ void gemm_bf16x3(const float* __restrict__ A,
                            const bf16_t* __restrict__ Bhi, const bf16_t* __restrict__ Blo,
                            const float* __restrict__ bias,
                            float* __restrict__ C, int M, int Ncols)
{
  const int lane  = threadIdx.x & 31;
  const int wave  = threadIdx.x >> 5;
  const int mTile = blockIdx.y * 8 + wave;
  if (mTile * 16 >= M) return;                 // wave-uniform: EXEC stays all-ones
  const int nBase = blockIdx.x * 128;

  const int l15  = lane & 15;
  const int hi16 = lane >> 4;
  const int kbA  = hi16 * 16;                  // A: 16 contiguous K per lane-half
  const int kbB  = hi16 * 16;                  // B: 16 contiguous K per lane-half
  const float* aRow = A + (size_t)(mTile * 16 + l15) * K;

  v8f acc[8] = {};

  for (int k0 = 0; k0 < K; k0 += 32) {
    bf16x16 ahi, alo;
    cvt_a_frag(aRow + k0 + kbA, ahi, alo);
#pragma unroll
    for (int nn = 0; nn < 8; ++nn) {
      const size_t bOff = (size_t)(nBase + nn * 16 + l15) * K + k0 + kbB;
      bf16x16 bhi = load_b_frag(Bhi + bOff);
      bf16x16 blo = load_b_frag(Blo + bOff);
      acc[nn] = __builtin_amdgcn_wmma_f32_16x16x32_bf16(false, ahi, false, bhi, (short)0, acc[nn], false, false);
      acc[nn] = __builtin_amdgcn_wmma_f32_16x16x32_bf16(false, ahi, false, blo, (short)0, acc[nn], false, false);
      acc[nn] = __builtin_amdgcn_wmma_f32_16x16x32_bf16(false, alo, false, bhi, (short)0, acc[nn], false, false);
    }
  }

  const int rowBase = mTile * 16 + hi16 * 8;   // D: elem v -> row v + 8*(lane>=16), col lane&15
#pragma unroll
  for (int nn = 0; nn < 8; ++nn) {
    const int col = nBase + nn * 16 + l15;
    const float b = bias ? bias[col] : 0.0f;
#pragma unroll
    for (int v = 0; v < 8; ++v)
      C[(size_t)(rowBase + v) * Ncols + col] = acc[nn][v] + b;
  }
}

// ---------- per-frame spatial attention (J=17, one wave per (frame, head)) ----------
__launch_bounds__(256)
__global__ void attn_kernel(const float* __restrict__ qkv,
                            const float* __restrict__ shared_attn,
                            const float* __restrict__ alpha_p,
                            float* __restrict__ y)
{
  __shared__ float sQ[8][JJ][33];   // padded stride 33 -> bank-conflict-free; reused for output
  __shared__ float sK[8][JJ][33];
  __shared__ float sV[8][JJ][33];

  const int lane = threadIdx.x & 31;
  const int w    = threadIdx.x >> 5;
  const int task = blockIdx.x * 8 + w;         // exactly 31104 tasks, no guard needed
  const int f    = task >> 3;                  // frame = b*T + t
  const int h    = task & 7;

  const size_t fbase = (size_t)f * JJ * QKVC;
  const int col = h * 32 + lane;               // lane = channel c here
#pragma unroll
  for (int j = 0; j < JJ; ++j) {
    const size_t rb = fbase + (size_t)j * QKVC;
    sQ[w][j][lane] = qkv[rb + col];
    sK[w][j][lane] = qkv[rb + INTERC + col];
    sV[w][j][lane] = qkv[rb + 2 * INTERC + col];
  }
  __syncthreads();

  const float alpha = alpha_p[0];
  if (lane < JJ) {                             // lane = joint row i
    float qr[32];
#pragma unroll
    for (int c = 0; c < 32; ++c) qr[c] = sQ[w][lane][c];
    float s[JJ];
    for (int jj = 0; jj < JJ; ++jj) {
      float a = 0.0f;
#pragma unroll
      for (int c = 0; c < 32; ++c) a += qr[c] * sK[w][jj][c];
      s[jj] = a * SCALE_F;
    }
    float m = s[0];
    for (int jj = 1; jj < JJ; ++jj) m = fmaxf(m, s[jj]);
    float sum = 0.0f;
    for (int jj = 0; jj < JJ; ++jj) { s[jj] = __expf(s[jj] - m); sum += s[jj]; }
    const float sc = alpha / sum;
    float attnv[JJ];
    for (int jj = 0; jj < JJ; ++jj)
      attnv[jj] = shared_attn[lane * JJ + jj] + s[jj] * sc;
    for (int c = 0; c < 32; ++c) {
      float o = 0.0f;
      for (int jj = 0; jj < JJ; ++jj) o += attnv[jj] * sV[w][jj][c];
      sQ[w][lane][c] = o;                      // reuse sQ as output staging
    }
  }
  __syncthreads();

#pragma unroll
  for (int j = 0; j < JJ; ++j) {               // coalesced writeout, lane = channel
    const size_t idx = ((size_t)f * JJ + j) * INTERC + h * 32 + lane;
    y[idx] = sQ[w][j][lane];
  }
}

// ---------- host launch ----------
extern "C" void kernel_launch(void* const* d_in, const int* in_sizes, int n_in,
                              void* d_out, int out_size, void* d_ws, size_t ws_size,
                              hipStream_t stream)
{
  const float* x           = (const float*)d_in[0];  // [66096, 512]
  const float* w_qkv       = (const float*)d_in[1];  // [768, 512]
  const float* w_proj      = (const float*)d_in[2];  // [512, 256]
  const float* b_proj      = (const float*)d_in[3];  // [512]
  const float* shared_attn = (const float*)d_in[4];  // [17,17]
  const float* alpha       = (const float*)d_in[5];  // [1]

  char* ws = (char*)d_ws;
  size_t off = 0;
  auto alloc = [&](size_t bytes) -> char* {
    char* p = ws + off;
    off += (bytes + 255) & ~(size_t)255;
    return p;
  };
  bf16_t* wq_hi = (bf16_t*)alloc((size_t)QKVC * DIMC * sizeof(bf16_t));
  bf16_t* wq_lo = (bf16_t*)alloc((size_t)QKVC * DIMC * sizeof(bf16_t));
  bf16_t* wp_hi = (bf16_t*)alloc((size_t)DIMC * INTERC * sizeof(bf16_t));
  bf16_t* wp_lo = (bf16_t*)alloc((size_t)DIMC * INTERC * sizeof(bf16_t));
  float*  qkv   = (float*)alloc((size_t)NROWS * QKVC * sizeof(float));   // ~203 MB
  float*  y     = (float*)alloc((size_t)NROWS * INTERC * sizeof(float)); // ~68 MB
  (void)ws_size; (void)in_sizes; (void)n_in; (void)out_size;

  // split weights into bf16 hi/lo (tiny)
  convert_hilo<<<192, 256, 0, stream>>>(w_qkv, wq_hi, wq_lo, QKVC * DIMC);
  convert_hilo<<<64, 256, 0, stream>>>(w_proj, wp_hi, wp_lo, DIMC * INTERC);

  // GEMM1: qkv = x @ w_qkv^T   (M=66096, K=512, N=768)
  {
    dim3 grid(QKVC / 128, (MTILES + 7) / 8);
    gemm_bf16x3<DIMC><<<grid, 256, 0, stream>>>(x, wq_hi, wq_lo, nullptr, qkv, NROWS, QKVC);
  }

  // attention: qkv -> y  (31104 wave-tasks, 8 per block)
  attn_kernel<<<(BATCH * TFRAMES * NHEADS) / 8, 256, 0, stream>>>(qkv, shared_attn, alpha, y);

  // GEMM2: out = y @ w_proj^T + b_proj   (M=66096, K=256, N=512)
  {
    dim3 grid(DIMC / 128, (MTILES + 7) / 8);
    gemm_bf16x3<INTERC><<<grid, 256, 0, stream>>>(y, wp_hi, wp_lo, b_proj, (float*)d_out, NROWS, DIMC);
  }
}
